// PositionwiseFeedforwardLayer_58334245814455
// MI455X (gfx1250) — matrix operand order repaired and checked by reading, not compiled, not run on hardware
//
#include <hip/hip_runtime.h>
#include <hip/hip_bf16.h>
#include <math.h>

// Problem constants (from reference): B=4, S=2048, H=1024, F=4096, E=8, TOP_K=2
#define T_TOK 8192
#define H_DIM 1024
#define F_DIM 4096
#define E_NUM 8
#define FC    256     // F-chunk processed per fused iteration
#define MT    16      // token tile (M) per block

typedef __attribute__((ext_vector_type(16))) __bf16 v16bf;
typedef __attribute__((ext_vector_type(8)))  float  v8f;

union PackBF2 { unsigned int u; __bf16 h[2]; };
union Frag32B { uint4 q[2]; v16bf v; };

// ---------------------------------------------------------------------------
// Fragment loaders
// ---------------------------------------------------------------------------
// Load a 32-byte WMMA fragment (16 bf16) from a 32B-aligned address: 2x b128.
__device__ __forceinline__ v16bf load_frag_b32B(const __bf16* p) {
  Frag32B u;
  const uint4* s = reinterpret_cast<const uint4*>(p);
  u.q[0] = s[0];
  u.q[1] = s[1];
  return u.v;
}

// B-fragment: 16 k-strided elements (table order, rows s + (s&8)) -> bf16.
__device__ __forceinline__ v16bf load_b_frag_f32(const float* p, size_t ld) {
  v16bf b;
#pragma unroll
  for (int s = 0; s < 16; ++s) b[s] = (__bf16)p[(size_t)(s + (s & 8)) * ld];
  return b;
}
__device__ __forceinline__ v16bf load_b_frag_f32(const __bf16* p, size_t ld) {
  v16bf b;
#pragma unroll
  for (int s = 0; s < 16; ++s) b[s] = p[(size_t)(s + (s & 8)) * ld];
  return b;
}

__device__ __forceinline__ v8f wmma_bf16(v16bf a, v16bf b, v8f c) {
  return __builtin_amdgcn_wmma_f32_16x16x32_bf16(false, a, false, b, (short)0, c,
                                                 false, false);
}

// Swizzled LDS layout for 16xK bf16 tiles, fragment-ready (ISA 7.12.2 A layout):
//   element (m, k) -> [kb = k>>5][lane = m + 16*((k>>3)&1)][slot = (k>>4 & 1)*8 + (k&7)]
// so a lane's A-fragment for K-block kb is 16 contiguous bf16 (one 32B read).
__device__ __forceinline__ int sw_idx(int m, int k) {
  int kl = k & 31;
  int lane = m + (((kl >> 3) & 1) << 4);
  int slot = ((kl >> 4) << 3) + (kl & 7);
  return (((k >> 5) << 5) + lane) * 16 + slot;
}

// ---------------------------------------------------------------------------
// Kernel 0: zero output accumulator + expert counters
// ---------------------------------------------------------------------------
__global__ void zero_init_kernel(float* __restrict__ y, int n,
                                 int* __restrict__ counts, int ec) {
  int i = blockIdx.x * blockDim.x + threadIdx.x;
  if (i < n) y[i] = 0.0f;
  if (i < ec) counts[i] = 0;
}

// ---------------------------------------------------------------------------
// Kernel 1: one-shot fp32 -> bf16 weight conversion (amortizes cvt across all
// tiles; bf16 weights for all 8 experts = 128 MB, resident in the 192 MB L2).
// ---------------------------------------------------------------------------
__global__ __launch_bounds__(256)
void convert_w_kernel(const float* __restrict__ src, __bf16* __restrict__ dst,
                      size_t n) {
  size_t i = ((size_t)blockIdx.x * 256 + threadIdx.x) * 8;
  if (i >= n) return;
  const float4* s = reinterpret_cast<const float4*>(src + i);
  float4 a = s[0], b = s[1];
  PackBF2 p0, p1, p2, p3;
  p0.h[0] = (__bf16)a.x; p0.h[1] = (__bf16)a.y;
  p1.h[0] = (__bf16)a.z; p1.h[1] = (__bf16)a.w;
  p2.h[0] = (__bf16)b.x; p2.h[1] = (__bf16)b.y;
  p3.h[0] = (__bf16)b.z; p3.h[1] = (__bf16)b.w;
  uint4 o = {p0.u, p1.u, p2.u, p3.u};
  *reinterpret_cast<uint4*>(dst + i) = o;  // global_store_b128
}

// ---------------------------------------------------------------------------
// Kernel 2: gating — softmax(x@wg), top-2 renorm, scatter to expert lists
// ---------------------------------------------------------------------------
__global__ __launch_bounds__(256)
void gate_kernel(const float* __restrict__ x, const float* __restrict__ wg,
                 int* __restrict__ counts, int* __restrict__ tok_list,
                 float* __restrict__ gate_list) {
  __shared__ float red[256 * E_NUM];
  const int t = blockIdx.x;
  const int tid = threadIdx.x;

  float acc[E_NUM];
#pragma unroll
  for (int e = 0; e < E_NUM; ++e) acc[e] = 0.0f;

  for (int h = tid; h < H_DIM; h += 256) {
    float xv = x[(size_t)t * H_DIM + h];
#pragma unroll
    for (int e = 0; e < E_NUM; ++e) acc[e] += xv * wg[h * E_NUM + e];
  }
#pragma unroll
  for (int e = 0; e < E_NUM; ++e) red[tid * E_NUM + e] = acc[e];
  __syncthreads();

  for (int s = 128; s > 0; s >>= 1) {
    if (tid < s) {
#pragma unroll
      for (int e = 0; e < E_NUM; ++e)
        red[tid * E_NUM + e] += red[(tid + s) * E_NUM + e];
    }
    __syncthreads();
  }

  if (tid == 0) {
    float p[E_NUM];
    float mx = red[0];
#pragma unroll
    for (int e = 1; e < E_NUM; ++e) mx = fmaxf(mx, red[e]);
    float sum = 0.0f;
#pragma unroll
    for (int e = 0; e < E_NUM; ++e) { p[e] = expf(red[e] - mx); sum += p[e]; }
#pragma unroll
    for (int e = 0; e < E_NUM; ++e) p[e] /= sum;

    int i0 = 0;
#pragma unroll
    for (int e = 1; e < E_NUM; ++e) if (p[e] > p[i0]) i0 = e;
    int i1 = (i0 == 0) ? 1 : 0;
#pragma unroll
    for (int e = 0; e < E_NUM; ++e) if (e != i0 && p[e] > p[i1]) i1 = e;

    float rs = 1.0f / (p[i0] + p[i1] + 1e-9f);
    float v0 = p[i0] * rs, v1 = p[i1] * rs;

    int pos0 = atomicAdd(&counts[i0], 1);
    tok_list[i0 * T_TOK + pos0]  = t;
    gate_list[i0 * T_TOK + pos0] = v0;
    int pos1 = atomicAdd(&counts[i1], 1);
    tok_list[i1 * T_TOK + pos1]  = t;
    gate_list[i1 * T_TOK + pos1] = v1;
  }
}

// ---------------------------------------------------------------------------
// Kernel 3: fused per-expert FFN on gathered tokens (bf16 WMMA).
// grid = (T/MT tiles, E experts), 256 threads = 8 waves.
//   GEMM1 (all waves): hdn = relu(x_tile @ w1[:,chunk] + b1)  -> swizzled LDS
//   GEMM2 (per wave):  yacc[128 cols] += hdn @ w2[chunk, cols]
// WBF16: weights pre-converted to bf16 (B-frag = 16 k-strided bf16 loads, no VALU), else
// fallback converts fp32 weights inline.
// ---------------------------------------------------------------------------
template <bool WBF16>
__global__ __launch_bounds__(256)
void moe_ffn_kernel(const float* __restrict__ x,
                    const float* __restrict__ w1f, const __bf16* __restrict__ w1b,
                    const float* __restrict__ b1,
                    const float* __restrict__ w2f, const __bf16* __restrict__ w2b,
                    const float* __restrict__ b2,
                    const int* __restrict__ counts,
                    const int* __restrict__ tok_list,
                    const float* __restrict__ gate_list,
                    float* __restrict__ y) {
  __shared__ __align__(32) __bf16 xs[MT * H_DIM];   // 32 KB, swizzled
  __shared__ __align__(32) __bf16 hdn[MT * FC];     //  8 KB, swizzled
  __shared__ int   s_tok[MT];
  __shared__ float s_gate[MT];

  const int e    = blockIdx.y;
  const int tile = blockIdx.x;
  const int cnt  = counts[e];
  const int row0 = tile * MT;
  if (row0 >= cnt) return;  // block-uniform early exit (EXEC stays all-ones)

  const int tid  = threadIdx.x;
  const int lane = tid & 31;
  const int wv   = tid >> 5;

  if (tid < MT) {
    int r = row0 + tid;
    if (r < cnt) {
      s_tok[tid]  = tok_list[e * T_TOK + r];
      s_gate[tid] = gate_list[e * T_TOK + r];
    } else {
      s_tok[tid]  = 0;       // padded row: gate 0 => contributes 0
      s_gate[tid] = 0.0f;
    }
  }
  __syncthreads();

  // Stage gathered x rows to swizzled LDS: pack f32 pairs -> one b32 store.
#pragma unroll
  for (int i = 0; i < (MT * H_DIM) / (2 * 256); ++i) {
    int p = tid + i * 256;          // pair index
    int r = p >> 9;                 // H_DIM/2 = 512 pairs per row
    int c = (p & 511) << 1;
    float2 f = make_float2(0.0f, 0.0f);
    if ((row0 + r) < cnt)
      f = *reinterpret_cast<const float2*>(x + (size_t)s_tok[r] * H_DIM + c);
    PackBF2 pk;
    pk.h[0] = (__bf16)f.x;
    pk.h[1] = (__bf16)f.y;
    reinterpret_cast<unsigned int*>(xs)[sw_idx(r, c) >> 1] = pk.u;
  }
  __syncthreads();

  const size_t wstride = (size_t)H_DIM * F_DIM;
  const float*  w1fe = w1f + (size_t)e * wstride;
  const float*  w2fe = w2f + (size_t)e * wstride;
  const __bf16* w1be = w1b + (size_t)e * wstride;
  const __bf16* w2be = w2b + (size_t)e * wstride;

  v8f yacc[8];
#pragma unroll
  for (int t = 0; t < 8; ++t) yacc[t] = v8f{};

  const int n  = lane & 15;
  const int mh = (lane >> 4) * 8;

  for (int fc0 = 0; fc0 < F_DIM; fc0 += FC) {
    // ---- GEMM1: this wave owns hdn column tiles 2*wv and 2*wv+1 ----
    const int fcolA = fc0 + (2 * wv) * 16;
    const int fcolB = fcolA + 16;
    v8f h0 = v8f{}, h1 = v8f{};
    for (int k0 = 0; k0 < H_DIM; k0 += 32) {
      v16bf a = load_frag_b32B(xs + (((k0 >> 5) << 5) + lane) * 16);
      v16bf bA, bB;
      if constexpr (WBF16) {
        const __bf16* wrow = w1be + (size_t)(k0 + mh) * F_DIM;
        __builtin_prefetch(wrow + 32 * F_DIM + fcolA, 0, 1);  // global_prefetch_b8
        bA = load_b_frag_f32(wrow + fcolA + n, F_DIM);
        bB = load_b_frag_f32(wrow + fcolB + n, F_DIM);
      } else {
        const float* wrow = w1fe + (size_t)(k0 + mh) * F_DIM;
        __builtin_prefetch(wrow + 32 * F_DIM + fcolA, 0, 1);
        bA = load_b_frag_f32(wrow + fcolA + n, F_DIM);
        bB = load_b_frag_f32(wrow + fcolB + n, F_DIM);
      }
      h0 = wmma_bf16(a, bA, h0);
      h1 = wmma_bf16(a, bB, h1);
    }
    {
      // bias + relu, write to swizzled hdn. Both tiles land in K-block wv:
      // lane' = m + 16*((n>>3)&1); tileA slot = n&7, tileB slot = 8 + (n&7).
      float bias0 = b1[e * F_DIM + fcolA + n];
      float bias1 = b1[e * F_DIM + fcolB + n];
      __bf16* hblk = hdn + (wv << 5) * 16;
      int laneh = (((n >> 3) & 1) << 4);
      int slotA = n & 7;
#pragma unroll
      for (int r = 0; r < 8; ++r) {
        float v0 = h0[r] + bias0; v0 = v0 > 0.0f ? v0 : 0.0f;
        float v1 = h1[r] + bias1; v1 = v1 > 0.0f ? v1 : 0.0f;
        int lp = (mh + r) + laneh;
        hblk[lp * 16 + slotA]     = (__bf16)v0;
        hblk[lp * 16 + slotA + 8] = (__bf16)v1;
      }
    }
    __syncthreads();  // hdn chunk visible to all waves

    // ---- GEMM2: this wave owns y column tiles {wv, wv+8, ..., wv+56} ----
    for (int k0 = 0; k0 < FC; k0 += 32) {
      v16bf a = load_frag_b32B(hdn + (((k0 >> 5) << 5) + lane) * 16);
#pragma unroll
      for (int t = 0; t < 8; ++t) {
        int ncol = (wv + 8 * t) * 16;
        v16bf b;
        if constexpr (WBF16) {
          const __bf16* wrow = w2be + (size_t)(fc0 + k0 + mh) * H_DIM;
          if (t == 0) __builtin_prefetch(wrow + 32 * H_DIM + wv * 16, 0, 1);
          b = load_b_frag_f32(wrow + ncol + n, H_DIM);
        } else {
          const float* wrow = w2fe + (size_t)(fc0 + k0 + mh) * H_DIM;
          if (t == 0) __builtin_prefetch(wrow + 32 * H_DIM + wv * 16, 0, 1);
          b = load_b_frag_f32(wrow + ncol + n, H_DIM);
        }
        yacc[t] = wmma_bf16(a, b, yacc[t]);
      }
    }
    __syncthreads();  // done reading hdn before next chunk overwrites it
  }

  // ---- Epilogue: +b2, scale by gate, scatter-accumulate into y ----
#pragma unroll
  for (int t = 0; t < 8; ++t) {
    int ncol = (wv + 8 * t) * 16 + n;
    float bias = b2[e * H_DIM + ncol];
#pragma unroll
    for (int r = 0; r < 8; ++r) {
      int row = mh + r;
      float val = s_gate[row] * (yacc[t][r] + bias);
      atomicAdd(&y[(size_t)s_tok[row] * H_DIM + ncol], val);  // f32 global atomic
    }
  }
}

// ---------------------------------------------------------------------------
// Host launcher
// ---------------------------------------------------------------------------
extern "C" void kernel_launch(void* const* d_in, const int* in_sizes, int n_in,
                              void* d_out, int out_size, void* d_ws, size_t ws_size,
                              hipStream_t stream) {
  const float* x  = (const float*)d_in[0];  // [B,S,H]
  const float* wg = (const float*)d_in[1];  // [H,E]
  const float* w1 = (const float*)d_in[2];  // [E,H,F]
  const float* b1 = (const float*)d_in[3];  // [E,F]
  const float* w2 = (const float*)d_in[4];  // [E,F,H]
  const float* b2 = (const float*)d_in[5];  // [E,H]
  float* y = (float*)d_out;                 // [B,S,H]

  // Workspace layout: [counters | token lists | gate lists] (1 MB region),
  // then optionally bf16 copies of w1 and w2 (64 MB each).
  char* ws = (char*)d_ws;
  int*   counts    = (int*)ws;                                   // E (padded)
  int*   tok_list  = (int*)ws + 64;                              // E*T
  float* gate_list = (float*)((int*)ws + 64 + E_NUM * T_TOK);    // E*T

  const size_t welems = (size_t)E_NUM * H_DIM * F_DIM;           // 32M each
  const size_t lists_bytes = (size_t)1 << 20;
  __bf16* w1b = (__bf16*)(ws + lists_bytes);
  __bf16* w2b = w1b + welems;
  const bool fast = ws_size >= lists_bytes + 2 * welems * sizeof(__bf16);

  const int n_out = T_TOK * H_DIM;
  zero_init_kernel<<<(n_out + 255) / 256, 256, 0, stream>>>(y, n_out, counts, E_NUM);
  gate_kernel<<<T_TOK, 256, 0, stream>>>(x, wg, counts, tok_list, gate_list);

  if (fast) {
    const int cblocks = (int)(welems / (8 * 256));
    convert_w_kernel<<<cblocks, 256, 0, stream>>>(w1, w1b, welems);
    convert_w_kernel<<<cblocks, 256, 0, stream>>>(w2, w2b, welems);
    moe_ffn_kernel<true><<<dim3(T_TOK / MT, E_NUM), 256, 0, stream>>>(
        x, w1, w1b, b1, w2, w2b, b2, counts, tok_list, gate_list, y);
  } else {
    moe_ffn_kernel<false><<<dim3(T_TOK / MT, E_NUM), 256, 0, stream>>>(
        x, w1, w1b, b1, w2, w2b, b2, counts, tok_list, gate_list, y);
  }
}
